// ROI_Relation_59365037965323
// MI455X (gfx1250) — compile-verified, matrix operand order checked by reading, not run on hardware
//
#include <hip/hip_runtime.h>
#include <cstdint>
#include <cstddef>

// Problem constants (match reference)
#define B_   8
#define T_   4096
#define C_   512
#define N_   512
#define P_   16
#define PT_  18
#define K_   (PT_ * C_)   // 9216
#define M_   (B_ * N_)    // 4096
#define EPS_ 1e-5f

typedef __bf16 v8bf  __attribute__((ext_vector_type(8)));
typedef __bf16 v16bf __attribute__((ext_vector_type(16)));
typedef float  v8f   __attribute__((ext_vector_type(8)));

// ---------------------------------------------------------------------------
// Phase 0: convert W (f32 [C_ x K_]) to bf16
// ---------------------------------------------------------------------------
__global__ void cvt_w_kernel(const float* __restrict__ W,
                             __bf16* __restrict__ Wb, int n) {
  int i = blockIdx.x * blockDim.x + threadIdx.x;
  int stride = gridDim.x * blockDim.x;
  for (; i < n; i += stride) Wb[i] = (__bf16)W[i];
}

// ---------------------------------------------------------------------------
// Phase 1: ROI pool (1 tap) + InstanceNorm over C, emit bf16 A row segment.
// One 256-thread block per (b, roi, pt); each thread owns 2 channels.
// ---------------------------------------------------------------------------
__global__ __launch_bounds__(256) void pool_norm_bf16_kernel(
    const float* __restrict__ feats,   // [B,T,C]
    const float* __restrict__ srois,   // [B,N,2]
    const float* __restrict__ erois,   // [B,N,2]
    const float* __restrict__ rois,    // [B,N,2]
    __bf16* __restrict__ A) {          // [M_, K_] bf16
  const int bnpt = blockIdx.x;
  const int pt = bnpt % PT_;
  const int bn = bnpt / PT_;           // = b*N_ + n
  const int b  = bn / N_;
  const int tid = threadIdx.x;

  float s, e, frac;
  if (pt == 0) {
    s = srois[bn * 2 + 0]; e = srois[bn * 2 + 1]; frac = 0.5f;
  } else if (pt == PT_ - 1) {
    s = erois[bn * 2 + 0]; e = erois[bn * 2 + 1]; frac = 0.5f;
  } else {
    s = rois[bn * 2 + 0];  e = rois[bn * 2 + 1];
    frac = ((float)(pt - 1) + 0.5f) / (float)P_;
  }
  float pos = s + (e - s) * frac;
  pos = fminf(fmaxf(pos, 0.0f), (float)(T_ - 1));
  int lo = (int)floorf(pos);
  int hi = min(lo + 1, T_ - 1);
  float w = pos - (float)lo;

  const float* f0 = feats + ((size_t)b * T_ + lo) * C_;
  const float* f1 = feats + ((size_t)b * T_ + hi) * C_;
  const int c0 = tid, c1 = tid + 256;
  float x0 = (1.0f - w) * f0[c0] + w * f1[c0];
  float x1 = (1.0f - w) * f0[c1] + w * f1[c1];

  // mean / var over 512 channels
  float sum = x0 + x1;
  float sq  = x0 * x0 + x1 * x1;
#pragma unroll
  for (int m = 16; m >= 1; m >>= 1) {
    sum += __shfl_xor(sum, m, 32);
    sq  += __shfl_xor(sq,  m, 32);
  }
  __shared__ float red[2][8];
  const int wid = tid >> 5;
  if ((tid & 31) == 0) { red[0][wid] = sum; red[1][wid] = sq; }
  __syncthreads();
  if (tid == 0) {
    float S = 0.f, Q = 0.f;
#pragma unroll
    for (int i = 0; i < 8; ++i) { S += red[0][i]; Q += red[1][i]; }
    float mean = S * (1.0f / (float)C_);
    float var  = Q * (1.0f / (float)C_) - mean * mean;
    red[0][0] = mean;
    red[1][0] = rsqrtf(var + EPS_);
  }
  __syncthreads();
  const float mean = red[0][0];
  const float rstd = red[1][0];

  __bf16* row = A + (size_t)bn * K_ + (size_t)pt * C_;
  row[c0] = (__bf16)((x0 - mean) * rstd);
  row[c1] = (__bf16)((x1 - mean) * rstd);
}

// ---------------------------------------------------------------------------
// Phase 2: bf16 WMMA GEMM [M_,K_] x [C_,K_]^T -> [M_,C_], + bias + SELU.
// Block tile 128x64, BK=64, 8 waves (4x2), wave tile 32x32 (2x2 WMMA frags).
// LDS fills via CDNA5 async global->LDS (ASYNCcnt), double buffered.
// ---------------------------------------------------------------------------
#define BM  128
#define BN  64
#define BK  64
#define LDA 72   // row padding: 8 bf16 (16 B) to spread LDS banks
#define LDB 72

__device__ __forceinline__ void async_b128_to_lds(uint32_t lds_off,
                                                  const void* gptr) {
  asm volatile("global_load_async_to_lds_b128 %0, %1, off"
               :: "v"(lds_off),
                  "v"((unsigned long long)(uintptr_t)gptr)
               : "memory");
}

__global__ __launch_bounds__(256) void gemm_selu_kernel(
    const __bf16* __restrict__ A,    // [M_, K_]
    const __bf16* __restrict__ Wb,   // [C_, K_]
    const float*  __restrict__ bias, // [C_]
    float* __restrict__ out) {       // [M_, C_]
  __shared__ __align__(16) __bf16 As[2][BM][LDA];
  __shared__ __align__(16) __bf16 Bs[2][BN][LDB];

  const int tid  = threadIdx.x;
  const int lane = tid & 31;
  const int wid  = tid >> 5;
  const int wm   = (wid >> 1) * 32;   // wave row offset in block tile
  const int wn   = (wid & 1) * 32;    // wave col offset
  const int row0 = blockIdx.x * BM;
  const int col0 = blockIdx.y * BN;

  const uint32_t as0 = (uint32_t)(uintptr_t)&As[0][0][0];
  const uint32_t bs0 = (uint32_t)(uintptr_t)&Bs[0][0][0];

  auto issue_tile = [&](int buf, int k0) {
    // A tile: 128x64 bf16 = 1024 x 16B chunks -> 4 per thread
#pragma unroll
    for (int i = 0; i < 4; ++i) {
      int idx = tid + i * 256;
      int r = idx >> 3;
      int c = (idx & 7) * 8;
      uint32_t loff = as0 + (uint32_t)((buf * BM + r) * LDA + c) * 2u;
      async_b128_to_lds(loff, A + (size_t)(row0 + r) * K_ + k0 + c);
    }
    // B tile: 64x64 bf16 = 512 x 16B chunks -> 2 per thread
#pragma unroll
    for (int i = 0; i < 2; ++i) {
      int idx = tid + i * 256;
      int r = idx >> 3;
      int c = (idx & 7) * 8;
      uint32_t loff = bs0 + (uint32_t)((buf * BN + r) * LDB + c) * 2u;
      async_b128_to_lds(loff, Wb + (size_t)(col0 + r) * K_ + k0 + c);
    }
  };

  v8f acc[2][2] = {};
  const int r16  = lane & 15;          // fragment row/col within 16
  const int koff = (lane >> 4) * 8;    // K sub-offset per ISA 16-bit layout

  issue_tile(0, 0);
  const int NKT = K_ / BK;             // 144
  for (int kt = 0; kt < NKT; ++kt) {
    const int buf = kt & 1;
    if (kt + 1 < NKT) {
      issue_tile(buf ^ 1, (kt + 1) * BK);
      asm volatile("s_wait_asynccnt %0" :: "n"(6) : "memory");
    } else {
      asm volatile("s_wait_asynccnt %0" :: "n"(0) : "memory");
    }
    __syncthreads();

#pragma unroll
    for (int kk = 0; kk < BK; kk += 32) {
      v16bf afr[2], bfr[2];
#pragma unroll
      for (int i = 0; i < 2; ++i) {
        const __bf16* p = &As[buf][wm + i * 16 + r16][kk + koff];
        v8bf lo = *(const v8bf*)p;
        v8bf hi = *(const v8bf*)(p + 16);
        afr[i] = __builtin_shufflevector(lo, hi, 0, 1, 2, 3, 4, 5, 6, 7,
                                         8, 9, 10, 11, 12, 13, 14, 15);
      }
#pragma unroll
      for (int j = 0; j < 2; ++j) {
        const __bf16* p = &Bs[buf][wn + j * 16 + r16][kk + koff];
        v8bf lo = *(const v8bf*)p;
        v8bf hi = *(const v8bf*)(p + 16);
        bfr[j] = __builtin_shufflevector(lo, hi, 0, 1, 2, 3, 4, 5, 6, 7,
                                         8, 9, 10, 11, 12, 13, 14, 15);
      }
#pragma unroll
      for (int i = 0; i < 2; ++i)
#pragma unroll
        for (int j = 0; j < 2; ++j)
          acc[i][j] = __builtin_amdgcn_wmma_f32_16x16x32_bf16(
              false, afr[i], false, bfr[j], (short)0, acc[i][j], false, false);
    }
    __syncthreads();
  }

  // Epilogue: bias + SELU, f32 out. D layout: VGPR v -> row v + 8*(lane/16),
  // col = lane%16 (per 16x16 f32 C/D layout).
  const float al = 1.6732632423543772f;
  const float sc = 1.0507009873554805f;
  const int mrow_base = row0 + wm + (lane >> 4) * 8;
#pragma unroll
  for (int j = 0; j < 2; ++j) {
    const int ncol = col0 + wn + j * 16 + r16;
    const float bv = bias[ncol];
#pragma unroll
    for (int i = 0; i < 2; ++i) {
#pragma unroll
      for (int v = 0; v < 8; ++v) {
        const int m = mrow_base + i * 16 + v;
        float x = acc[i][j][v] + bv;
        float y = (x > 0.0f) ? sc * x : sc * al * (__expf(x) - 1.0f);
        out[(size_t)m * C_ + ncol] = y;
      }
    }
  }
}

// ---------------------------------------------------------------------------
extern "C" void kernel_launch(void* const* d_in, const int* in_sizes, int n_in,
                              void* d_out, int out_size, void* d_ws,
                              size_t ws_size, hipStream_t stream) {
  (void)in_sizes; (void)n_in; (void)out_size; (void)ws_size;
  const float* feats = (const float*)d_in[0];  // [B,T,C]
  const float* srois = (const float*)d_in[1];  // [B,N,2]
  const float* erois = (const float*)d_in[2];  // [B,N,2]
  const float* rois  = (const float*)d_in[3];  // [B,N,2]
  // d_in[4] rois_mask, d_in[5] rois_pos_emb: unused by the reference
  const float* W     = (const float*)d_in[6];  // [C, PT*C]
  const float* bias  = (const float*)d_in[7];  // [C]
  float* out = (float*)d_out;                  // [B,N,C] f32

  __bf16* Abf = (__bf16*)d_ws;                                   // M_ * K_
  __bf16* Wbf = (__bf16*)((char*)d_ws + (size_t)M_ * K_ * 2);    // C_ * K_

  cvt_w_kernel<<<4096, 256, 0, stream>>>(W, Wbf, C_ * K_);
  pool_norm_bf16_kernel<<<B_ * N_ * PT_, 256, 0, stream>>>(
      feats, srois, erois, rois, Abf);
  dim3 grid(M_ / BM, C_ / BN);
  gemm_selu_kernel<<<grid, 256, 0, stream>>>(Abf, Wbf, bias, out);
}
